// LSTM_46291157517096
// MI455X (gfx1250) — matrix operand order repaired and checked by reading, not compiled, not run on hardware
//
#include <hip/hip_runtime.h>
#include <hip/hip_bf16.h>

// ---------------------------------------------------------------------------
// LSTM (B=128, S=512, H=512, C=10) for gfx1250 (MI455X).
//
//  * Batch rows independent -> 8 persistent workgroups x 16 rows, no grid sync.
//  * 32 waves/WG; wave w owns hidden columns [16w,16w+16) for all 4 gates and
//    the cell state (VGPR-resident across all 512 steps).
//  * v_wmma_f32_16x16x32_bf16; A = h tile from LDS, B = bf16 weights packed
//    into WMMA fragment layout, streamed from L2 (2 MB, L2-resident).
//  * B fragments: distance-1 register pipeline (+ prefetches that double as
//    scheduler separators) -> WMMAs wait on loads issued a full iteration
//    earlier (s_wait_loadcnt 0x6), ~32 KB in flight per SIMD.
//  * h double-buffered in LDS -> single s_barrier per timestep.
// ---------------------------------------------------------------------------

typedef __attribute__((ext_vector_type(16))) __bf16 v16bf;
typedef __attribute__((ext_vector_type(8)))  __bf16 v8bf;
typedef __attribute__((ext_vector_type(8)))  float  v8f;

#define HDIM 512
#define SDIM 512
#define BDIM 128
#define CDIM 10

union AU {
    v16bf v;
    struct { v8bf lo; v8bf hi; } s;
};

__device__ __forceinline__ float sigmoid_f(float x) {
    return __builtin_amdgcn_rcpf(1.0f + __expf(-x));
}
__device__ __forceinline__ float tanh_f(float x) {
    return 2.0f * __builtin_amdgcn_rcpf(1.0f + __expf(-2.0f * x)) - 1.0f;
}

#define WMMA_BF16(A, B, C) \
    __builtin_amdgcn_wmma_f32_16x16x32_bf16(false, (A), false, (B), (short)0, (C), false, false)

// ---------------------------------------------------------------------------
// Pack the four HxH fp32 recurrent weight matrices into bf16 WMMA B-fragment
// layout: element index = (((g*32 + ntile)*16 + ks)*32 + lane)*16 + j
//   B[k][n] = W[n][k],  n = ntile*16 + (lane&15),
//   k = ks*32 + (lane<16 ? 0 : 16) + j      (j = 0..15, contiguous K chunk)
// ---------------------------------------------------------------------------
__global__ void pack_weights_kernel(const float* __restrict__ Wg,
                                    const float* __restrict__ Wi,
                                    const float* __restrict__ Wf,
                                    const float* __restrict__ Wo,
                                    __bf16* __restrict__ out) {
    int idx = blockIdx.x * blockDim.x + threadIdx.x;   // 0 .. 4*512*512-1
    int j     =  idx        & 15;
    int lane  = (idx >> 4)  & 31;
    int ks    = (idx >> 9)  & 15;
    int ntile = (idx >> 13) & 31;
    int g     = (idx >> 18) & 3;
    int n = ntile * 16 + (lane & 15);
    int k = ks * 32 + ((lane < 16) ? 0 : 16) + j;
    const float* W = (g == 0) ? Wg : (g == 1) ? Wi : (g == 2) ? Wf : Wo;
    out[idx] = (__bf16)W[n * HDIM + k];
}

// ---------------------------------------------------------------------------
// Persistent LSTM kernel: one workgroup = 16 batch rows, 1024 threads.
// ---------------------------------------------------------------------------
__global__ __launch_bounds__(1024, 1)
void lstm_kernel(const float* __restrict__ x,
                 const float* __restrict__ Wgx, const float* __restrict__ bg,
                 const float* __restrict__ Wix, const float* __restrict__ bi,
                 const float* __restrict__ Wfx, const float* __restrict__ bfv_,
                 const float* __restrict__ Wox, const float* __restrict__ bo,
                 const float* __restrict__ Wph, const float* __restrict__ bp,
                 const __bf16* __restrict__ wpk,
                 float* __restrict__ out) {
    __shared__ __bf16 hbuf[2][16 * HDIM];   // 2 x 16KB double-buffered h (bf16)
    __shared__ float  xsh[2][16];           // staged x_t for the 16 rows

    const int tid     = threadIdx.x;
    const int lane    = tid & 31;
    const int w       = tid >> 5;           // wave id 0..31 -> N columns
    const int rowbase = blockIdx.x * 16;    // batch rows handled by this WG
    const int n       = w * 16 + (lane & 15);       // owned N column (C layout)
    const int hi8     = (lane < 16) ? 0 : 8;        // C-layout row offset
    const int ma      = lane & 15;                  // A-layout row

    // h0 = 0
    for (int i = tid; i < 16 * HDIM; i += 1024) hbuf[0][i] = (__bf16)0.0f;

    // Per-lane invariant x-weights / biases for owned column n.
    const float wxg = Wgx[n], wxi = Wix[n], wxf = Wfx[n], wxo = Wox[n];
    const float bgv = bg[n],  biv = bi[n],  bfv = bfv_[n], bov = bo[n];

    // Packed-weight base pointers for this wave (per gate); stride 512 per ks.
    const __bf16* pg = wpk + (size_t)(0 * 32 + w) * 8192 + lane * 16;
    const __bf16* pi = wpk + (size_t)(1 * 32 + w) * 8192 + lane * 16;
    const __bf16* pf = wpk + (size_t)(2 * 32 + w) * 8192 + lane * 16;
    const __bf16* po = wpk + (size_t)(3 * 32 + w) * 8192 + lane * 16;

    // Cell state: 16x16 tile fragment, persistent in VGPRs.
    float cc[8];
#pragma unroll
    for (int r = 0; r < 8; ++r) cc[r] = 0.0f;

    int cur = 0;
    for (int t = 0; t < SDIM; ++t) {
        if (tid < 16) xsh[t & 1][tid] = x[(rowbase + tid) * SDIM + t];
        __syncthreads();   // xsh ready; previous step's h writes visible

        const __bf16* hc = hbuf[cur];

        // Accumulator init: rank-1 x-term + bias.
        float xrow[8];
#pragma unroll
        for (int r = 0; r < 8; ++r) xrow[r] = xsh[t & 1][r + hi8];
        v8f ag, ai, af, ao;
#pragma unroll
        for (int r = 0; r < 8; ++r) {
            ag[r] = xrow[r] * wxg + bgv;
            ai[r] = xrow[r] * wxi + biv;
            af[r] = xrow[r] * wxf + bfv;
            ao[r] = xrow[r] * wxo + bov;
        }

        // -------- K loop: reg pipeline (d=1) + prefetch separators ----------
        v16bf bgc = *(const v16bf*)(pg);
        v16bf bic = *(const v16bf*)(pi);
        v16bf bfc = *(const v16bf*)(pf);
        v16bf boc = *(const v16bf*)(po);

#pragma unroll 4
        for (int ks = 0; ks < 16; ++ks) {
            // Prefetch ks+3 (wraps: pre-warms next timestep's first tiles).
            const int kp = (ks + 3) & 15;
            __builtin_prefetch((const void*)(pg + kp * 512), 0, 3);
            __builtin_prefetch((const void*)(pi + kp * 512), 0, 3);
            __builtin_prefetch((const void*)(pf + kp * 512), 0, 3);
            __builtin_prefetch((const void*)(po + kp * 512), 0, 3);

            // Demand-load next iteration's B fragments.
            const int kn = (ks + 1) & 15;
            v16bf bgn = *(const v16bf*)(pg + kn * 512);
            v16bf bin = *(const v16bf*)(pi + kn * 512);
            v16bf bfn = *(const v16bf*)(pf + kn * 512);
            v16bf bon = *(const v16bf*)(po + kn * 512);

            // A fragment from LDS (low latency).
            AU au;
            const v8bf* pa = (const v8bf*)(hc + ma * HDIM + ks * 32 + 2 * hi8);
            au.s.lo = pa[0];   // K = base .. base+7
            au.s.hi = pa[1];   // K = base+8 .. base+15
            const v16bf a = au.v;

            // WMMAs consume fragments loaded one iteration ago.
            ag = WMMA_BF16(a, bgc, ag);
            ai = WMMA_BF16(a, bic, ai);
            af = WMMA_BF16(a, bfc, af);
            ao = WMMA_BF16(a, boc, ao);

            bgc = bgn; bic = bin; bfc = bfn; boc = bon;
        }

        // Activations + cell update, write h_new tile into the other buffer.
        __bf16* hn = hbuf[cur ^ 1];
#pragma unroll
        for (int r = 0; r < 8; ++r) {
            float gv = tanh_f(ag[r]);
            float iv = sigmoid_f(ai[r]);
            float fv = sigmoid_f(af[r]);
            float ov = sigmoid_f(ao[r]);
            float cn = gv * iv + cc[r] * fv;
            cc[r] = cn;
            float hv = tanh_f(cn) * ov;
            hn[(r + hi8) * HDIM + n] = (__bf16)hv;
        }
        cur ^= 1;
    }
    __syncthreads();

    // Final projection: out[row][cls] = h_T[row] . Wph[cls] + bp[cls]
    if (tid < 16 * CDIM) {
        const int row = tid / CDIM;
        const int cls = tid % CDIM;
        const __bf16* hr = hbuf[cur] + row * HDIM;
        float s = bp[cls];
        for (int k = 0; k < HDIM; ++k)
            s += (float)hr[k] * Wph[cls * HDIM + k];
        out[(rowbase + row) * CDIM + cls] = s;
    }
}

// ---------------------------------------------------------------------------
extern "C" void kernel_launch(void* const* d_in, const int* in_sizes, int n_in,
                              void* d_out, int out_size, void* d_ws, size_t ws_size,
                              hipStream_t stream) {
    const float* x   = (const float*)d_in[0];
    const float* Wgx = (const float*)d_in[1];
    const float* Wgh = (const float*)d_in[2];
    const float* bg  = (const float*)d_in[3];
    const float* Wix = (const float*)d_in[4];
    const float* Wih = (const float*)d_in[5];
    const float* bi  = (const float*)d_in[6];
    const float* Wfx = (const float*)d_in[7];
    const float* Wfh = (const float*)d_in[8];
    const float* bf  = (const float*)d_in[9];
    const float* Wox = (const float*)d_in[10];
    const float* Woh = (const float*)d_in[11];
    const float* bo  = (const float*)d_in[12];
    const float* Wph = (const float*)d_in[13];
    const float* bp  = (const float*)d_in[14];

    __bf16* wpk = (__bf16*)d_ws;   // 4*512*512 bf16 = 2 MB packed weights

    // One-time weight conversion + WMMA-fragment packing (stream-ordered).
    pack_weights_kernel<<<4096, 256, 0, stream>>>(Wgh, Wih, Wfh, Woh, wpk);

    // Persistent recurrence: 8 WGs x 1024 threads (16 batch rows each).
    lstm_kernel<<<8, 1024, 0, stream>>>(x, Wgx, bg, Wix, bi, Wfx, bf,
                                        Wox, bo, Wph, bp, wpk, (float*)d_out);
}
